// LSTM_36627481101171
// MI455X (gfx1250) — matrix operand order repaired and checked by reading, not compiled, not run on hardware
//
#include <hip/hip_runtime.h>
#include <math.h>

// Problem dims (fixed by the reference)
constexpr int N  = 64;
constexpr int T  = 256;
constexpr int I  = 512;
constexpr int H  = 1024;
constexpr int G  = 4 * H;        // 4096 gate columns
constexpr int KC = I + H;        // 1536 concatenated K
constexpr int NTH = N * T * H;   // per-output element count

// ---------------- WMMA types ----------------
typedef __attribute__((ext_vector_type(16))) __bf16        v16bf;
typedef __attribute__((ext_vector_type(8)))  float         v8f;
typedef __attribute__((ext_vector_type(4)))  unsigned int  v4u;

union Frag { v16bf v; v4u q[2]; };

// A-matrix 16x32 bf16 fragment (per ISA 7.12.2): lane = kh*16 + m,
// holds K = k0 + kh*8 + {0..7} and k0 + 16 + kh*8 + {0..7} of row m.
__device__ __forceinline__ void load_fragA(Frag& f, const unsigned short* rowp,
                                           int k0, int kh) {
  f.q[0] = *(const v4u*)(rowp + k0 + kh * 8);
  f.q[1] = *(const v4u*)(rowp + k0 + 16 + kh * 8);
}

// B-matrix 32x16 bf16 fragment: lane = kh*16 + n, holds K = k0 + kh*8 + {0..7}
// and k0 + 16 + kh*8 + {0..7} of weight row n (W stored row-major over K).
__device__ __forceinline__ void load_fragB(Frag& f, const unsigned short* rowp,
                                           int k0, int kh) {
  const unsigned short* p = rowp + k0 + kh * 8;
  f.q[0] = *(const v4u*)(p);
  f.q[1] = *(const v4u*)(p + 16);
}

__device__ __forceinline__ unsigned short f32_to_bf16(float x) {
  unsigned int u = __float_as_uint(x);
  if ((u & 0x7F800000u) == 0x7F800000u)  // Inf/NaN: truncate
    return (unsigned short)(u >> 16);
  u += 0x7FFFu + ((u >> 16) & 1u);       // round-to-nearest-even
  return (unsigned short)(u >> 16);
}

__device__ __forceinline__ float sigmoidf_(float x) {
  return 1.0f / (1.0f + expf(-x));
}
__device__ __forceinline__ float mishf_(float x) {
  float sp = (x > 20.0f) ? x : log1pf(expf(x));
  return x * tanhf(sp);
}

// ---------------- Prep kernels ----------------
__global__ __launch_bounds__(256) void k_cvt_bf16(const float* __restrict__ src,
                                                  unsigned short* __restrict__ dst,
                                                  int count) {
  int idx = blockIdx.x * 256 + threadIdx.x;
  if (idx < count) dst[idx] = f32_to_bf16(src[idx]);
}

// Wcat[r][0:I] = W_ih[r], Wcat[r][I:I+H] = W_hh[r]  (bf16, row-major over K)
__global__ __launch_bounds__(256) void k_wcat(const float* __restrict__ W_ih,
                                              const float* __restrict__ W_hh,
                                              unsigned short* __restrict__ Wcat) {
  int idx = blockIdx.x * 256 + threadIdx.x;          // G * KC threads
  int row = idx / KC;
  int col = idx - row * KC;
  float v = (col < I) ? W_ih[(size_t)row * I + col]
                      : W_hh[(size_t)row * H + (col - I)];
  Wcat[idx] = f32_to_bf16(v);
}

__global__ __launch_bounds__(256) void k_bias(const float* __restrict__ b_ih,
                                              const float* __restrict__ b_hh,
                                              float* __restrict__ bias) {
  int idx = blockIdx.x * 256 + threadIdx.x;          // G threads
  bias[idx] = b_ih[idx] + b_hh[idx];
}

// h0 = hx[:,0], c0 = cx[:,0]
__global__ __launch_bounds__(256) void k_init_state(const float* __restrict__ hx,
                                                    const float* __restrict__ cx,
                                                    float* __restrict__ h_state,
                                                    float* __restrict__ c_state,
                                                    unsigned short* __restrict__ h_bf) {
  int idx = blockIdx.x * 256 + threadIdx.x;          // N*H threads
  int n = idx >> 10;
  int k = idx & (H - 1);
  float h = hx[(size_t)n * T * H + k];
  float c = cx[(size_t)n * T * H + k];
  h_state[idx] = h;
  c_state[idx] = c;
  h_bf[idx] = f32_to_bf16(h);
}

// ---------------- Per-step gates GEMM (WMMA bf16) ----------------
// gates[64][4096] = [x_t | h*r] @ Wcat^T + bias ; 1024 tiles, 1 tile/wave.
__global__ __launch_bounds__(256) void k_gates(const unsigned short* __restrict__ x_bf,
                                               const unsigned short* __restrict__ h_bf,
                                               const unsigned short* __restrict__ Wcat,
                                               const float* __restrict__ bias,
                                               const int* __restrict__ is_init,
                                               float* __restrict__ gates, int t) {
  int tid  = threadIdx.x;
  int lane = tid & 31;
  int wave = blockIdx.x * 8 + (tid >> 5);      // 128 blocks * 8 waves = 1024
  int n_tile = wave & 255;                     // 256 tiles along G
  int m_tile = wave >> 8;                      // 4 tiles along N
  int m0 = m_tile * 16, n0 = n_tile * 16;
  int r  = lane & 15;
  int kh = lane >> 4;

  int mrow = m0 + r;                           // batch row for A lanes
  const unsigned short* arow_x = x_bf + ((size_t)mrow * T + t) * I;
  const unsigned short* arow_h = h_bf + (size_t)mrow * H;
  const unsigned short* brow   = Wcat + (size_t)(n0 + r) * KC;
  bool reset = (is_init[mrow * T + t] != 0);

  v8f acc = {};
  Frag a, b;

  // K over x portion
  for (int k0 = 0; k0 < I; k0 += 32) {
    load_fragA(a, arow_x, k0, kh);
    load_fragB(b, brow, k0, kh);
    acc = __builtin_amdgcn_wmma_f32_16x16x32_bf16(false, a.v, false, b.v,
                                                  (short)0, acc, false, false);
  }
  // K over h portion (apply per-row reset by zeroing the fragment)
  for (int k0 = 0; k0 < H; k0 += 32) {
    if (reset) {
      v4u z = {0u, 0u, 0u, 0u};
      a.q[0] = z; a.q[1] = z;
    } else {
      load_fragA(a, arow_h, k0, kh);
    }
    load_fragB(b, brow, I + k0, kh);
    acc = __builtin_amdgcn_wmma_f32_16x16x32_bf16(false, a.v, false, b.v,
                                                  (short)0, acc, false, false);
  }

  // C/D layout: lane -> col = n0 + (lane&15); VGPR j -> row m0 + (lane>>4)*8 + j
  int col   = n0 + r;
  float bc  = bias[col];
  int mbase = m0 + kh * 8;
  for (int j = 0; j < 8; ++j)
    gates[(size_t)(mbase + j) * G + col] = acc[j] + bc;
}

// ---------------- Per-step cell update ----------------
__global__ __launch_bounds__(256) void k_cell(const float* __restrict__ gates,
                                              const int* __restrict__ is_init,
                                              float* __restrict__ h_state,
                                              float* __restrict__ c_state,
                                              unsigned short* __restrict__ h_bf,
                                              unsigned short* __restrict__ hs_bf,
                                              int t) {
  int idx = blockIdx.x * 256 + threadIdx.x;    // N*H threads
  int n = idx >> 10;
  int k = idx & (H - 1);
  float rm = (is_init[n * T + t] != 0) ? 0.0f : 1.0f;
  const float* g = gates + (size_t)n * G;
  float ig = g[k], fg = g[H + k], gg = g[2 * H + k], og = g[3 * H + k];
  float c  = c_state[idx] * rm;
  float cn = sigmoidf_(fg) * c + sigmoidf_(ig) * tanhf(gg);
  float hn = sigmoidf_(og) * tanhf(cn);
  c_state[idx] = cn;
  h_state[idx] = hn;
  unsigned short hb = f32_to_bf16(hn);
  h_bf[idx] = hb;
  hs_bf[((size_t)n * T + t) * H + k] = hb;
}

// ---------------- Output projection: out = mish(hs @ W_out^T + b_out) ----------------
// M = N*T = 16384, K = H, N' = H. Each wave: 1 M-tile x 4 N-tiles (A reused 4x).
__global__ __launch_bounds__(256) void k_proj(const unsigned short* __restrict__ hs_bf,
                                              const unsigned short* __restrict__ Wout_bf,
                                              const float* __restrict__ b_out,
                                              float* __restrict__ out) {
  int tid  = threadIdx.x;
  int lane = tid & 31;
  int wave = blockIdx.x * 8 + (tid >> 5);      // 2048 blocks * 8 = 16384 waves
  int ngrp   = wave & 15;                      // 16 groups of 64 cols
  int m_tile = wave >> 4;                      // 1024 M-tiles
  int m0 = m_tile * 16, n0 = ngrp * 64;
  int r  = lane & 15;
  int kh = lane >> 4;

  const unsigned short* arow = hs_bf + (size_t)(m0 + r) * H;
  const unsigned short* br0  = Wout_bf + (size_t)(n0 + r) * H;

  v8f a0 = {}, a1 = {}, a2 = {}, a3 = {};
  Frag a, b0, b1, b2, b3;
  for (int k0 = 0; k0 < H; k0 += 32) {
    load_fragA(a, arow, k0, kh);
    load_fragB(b0, br0,            k0, kh);
    load_fragB(b1, br0 + 16 * H,   k0, kh);
    load_fragB(b2, br0 + 32 * H,   k0, kh);
    load_fragB(b3, br0 + 48 * H,   k0, kh);
    a0 = __builtin_amdgcn_wmma_f32_16x16x32_bf16(false, a.v, false, b0.v, (short)0, a0, false, false);
    a1 = __builtin_amdgcn_wmma_f32_16x16x32_bf16(false, a.v, false, b1.v, (short)0, a1, false, false);
    a2 = __builtin_amdgcn_wmma_f32_16x16x32_bf16(false, a.v, false, b2.v, (short)0, a2, false, false);
    a3 = __builtin_amdgcn_wmma_f32_16x16x32_bf16(false, a.v, false, b3.v, (short)0, a3, false, false);
  }

  int mbase = m0 + kh * 8;
  v8f accs[4] = {a0, a1, a2, a3};
  for (int ti = 0; ti < 4; ++ti) {
    int col  = n0 + ti * 16 + r;
    float bc = b_out[col];
    for (int j = 0; j < 8; ++j) {
      float v = accs[ti][j] + bc;
      out[(size_t)(mbase + j) * H + col] = mishf_(v);
    }
  }
}

// ---------------- Broadcast final h/c ----------------
__global__ __launch_bounds__(256) void k_finals(const float* __restrict__ h_state,
                                                const float* __restrict__ c_state,
                                                float* __restrict__ out1,
                                                float* __restrict__ out2) {
  int idx = blockIdx.x * 256 + threadIdx.x;    // N*T*H threads
  int n = idx >> 18;                           // / (T*H)
  int k = idx & (H - 1);
  float h = h_state[n * H + k];
  float c = c_state[n * H + k];
  out1[idx] = h;
  out2[idx] = c;
}

// ---------------- Workspace layout (bytes, 256-aligned) ----------------
constexpr size_t OFF_XBF   = 0;                         // N*T*I bf16      = 16 MB
constexpr size_t OFF_WCAT  = OFF_XBF  + (size_t)N*T*I*2;      // G*KC bf16 = 12 MB
constexpr size_t OFF_WOUT  = OFF_WCAT + (size_t)G*KC*2;       // H*H bf16  =  2 MB
constexpr size_t OFF_HSBF  = OFF_WOUT + (size_t)H*H*2;        // N*T*H bf16= 32 MB
constexpr size_t OFF_GATES = OFF_HSBF + (size_t)N*T*H*2;      // N*G f32   =  1 MB
constexpr size_t OFF_HST   = OFF_GATES+ (size_t)N*G*4;        // N*H f32
constexpr size_t OFF_CST   = OFF_HST  + (size_t)N*H*4;        // N*H f32
constexpr size_t OFF_HBF   = OFF_CST  + (size_t)N*H*4;        // N*H bf16
constexpr size_t OFF_BIAS  = OFF_HBF  + (size_t)N*H*2;        // G f32
// total ~66.8 MB

extern "C" void kernel_launch(void* const* d_in, const int* in_sizes, int n_in,
                              void* d_out, int out_size, void* d_ws, size_t ws_size,
                              hipStream_t stream) {
  (void)in_sizes; (void)n_in; (void)out_size; (void)ws_size;
  const float* x       = (const float*)d_in[0];
  const int*   is_init = (const int*)  d_in[1];
  const float* hx      = (const float*)d_in[2];
  const float* cx      = (const float*)d_in[3];
  const float* W_ih    = (const float*)d_in[4];
  const float* W_hh    = (const float*)d_in[5];
  const float* b_ih    = (const float*)d_in[6];
  const float* b_hh    = (const float*)d_in[7];
  const float* W_out   = (const float*)d_in[8];
  const float* b_out   = (const float*)d_in[9];
  float* out = (float*)d_out;

  char* ws = (char*)d_ws;
  unsigned short* x_bf    = (unsigned short*)(ws + OFF_XBF);
  unsigned short* Wcat    = (unsigned short*)(ws + OFF_WCAT);
  unsigned short* Wout_bf = (unsigned short*)(ws + OFF_WOUT);
  unsigned short* hs_bf   = (unsigned short*)(ws + OFF_HSBF);
  float*          gates   = (float*)(ws + OFF_GATES);
  float*          h_state = (float*)(ws + OFF_HST);
  float*          c_state = (float*)(ws + OFF_CST);
  unsigned short* h_bf    = (unsigned short*)(ws + OFF_HBF);
  float*          bias    = (float*)(ws + OFF_BIAS);

  // Prep: bf16 conversions, concatenated weights, combined bias, initial state.
  k_cvt_bf16 <<<(N*T*I)/256, 256, 0, stream>>>(x, x_bf, N*T*I);
  k_cvt_bf16 <<<(H*H)/256,   256, 0, stream>>>(W_out, Wout_bf, H*H);
  k_wcat     <<<(G*KC)/256,  256, 0, stream>>>(W_ih, W_hh, Wcat);
  k_bias     <<<G/256,       256, 0, stream>>>(b_ih, b_hh, bias);
  k_init_state<<<(N*H)/256,  256, 0, stream>>>(hx, cx, h_state, c_state, h_bf);

  // Serial recurrence: stream ordering carries the h/c dependency.
  for (int t = 0; t < T; ++t) {
    k_gates<<<128, 256, 0, stream>>>(x_bf, h_bf, Wcat, bias, is_init, gates, t);
    k_cell <<<(N*H)/256, 256, 0, stream>>>(gates, is_init, h_state, c_state,
                                           h_bf, hs_bf, t);
  }

  // Parallel output projection + mish, then broadcast finals.
  k_proj  <<<2048, 256, 0, stream>>>(hs_bf, Wout_bf, b_out, out);
  k_finals<<<NTH/256, 256, 0, stream>>>(h_state, c_state, out + NTH, out + 2*NTH);
}
